// PointNetSetAbstraction_15788299780823
// MI455X (gfx1250) — matrix operand order repaired and checked by reading, not compiled, not run on hardware
//
#include <hip/hip_runtime.h>
#include <hip/hip_bf16.h>

typedef _Float16 half_t;
typedef __attribute__((ext_vector_type(16))) _Float16 v16h;
typedef __attribute__((ext_vector_type(8)))  float    v8f;

#define BB   16
#define NN   4096
#define SS   1024
#define KNS  32
#define DD   64
#define CINP 70
#define MLP0C 64
#define MLP1C 128
#define EPSBN 1e-5f

static __device__ __forceinline__ v8f vzero8() {
  v8f v;
#pragma unroll
  for (int i = 0; i < 8; ++i) v[i] = 0.f;
  return v;
}

static __device__ __forceinline__ v8f wmma16(v16h a, v16h b, v8f c) {
  // D = A(16x32 f16) * B(32x16 f16) + C(16x16 f32)
  return __builtin_amdgcn_wmma_f32_16x16x32_f16(false, a, false, b, (short)0, c,
                                                false, false);
}

// Build an A-fragment (16x32 f16 tile) from LDS, contiguous K order:
// lanes 0-15: row M=lane, K 0..15
// lanes 16-31: same rows, K 16..31
static __device__ __forceinline__ v16h load_a_frag(const half_t* lds, int ld,
                                                   int mbase, int kt) {
  int lane = threadIdx.x & 31;
  int m = mbase + (lane & 15);
  int koff = kt * 32 + ((lane >> 4) << 4);
  const half_t* p = lds + m * ld + koff;
  v16h a;
#pragma unroll
  for (int v = 0; v < 4; ++v) {
    a[2 * v]     = p[2 * v];
    a[2 * v + 1] = p[2 * v + 1];
    a[2 * v + 8] = p[8 + 2 * v];
    a[2 * v + 9] = p[8 + 2 * v + 1];
  }
  return a;
}

// B fragments are pre-packed in workspace: frag (kt,nt) stored as 32 lanes x 16
// halves contiguous: lane holds column n = nt*16+(lane&15), K-chunk
// kt*32 + (lane>=16 ? 16 : 0) + e  (e = 0..15, contiguous K per lane half).
static __device__ __forceinline__ v16h load_b_frag(const half_t* base, int NT,
                                                   int kt, int nt) {
  int lane = threadIdx.x & 31;
  return *(const v16h*)(base + ((size_t)(kt * NT + nt) * 32 + lane) * 16);
}

// ---------------------------------------------------------------- weight prep
__global__ void __launch_bounds__(256) prep_frags_kernel(
    const float* Wf1, const float* gf1, const float* Wf2, const float* gf2,
    const float* Wp1, const float* gp1, const float* Wp2, const float* gp2,
    const float* Wkf, const float* Wvf, const float* Wkp, const float* Wvp,
    half_t* dWf1, half_t* dWf2, half_t* dWp1, half_t* dWp2,
    half_t* dWkf, half_t* dWvf, half_t* dWkp, half_t* dWvp) {
  const float* W; const float* g; int Nout, Kin, KT, NT; half_t* dst;
  switch (blockIdx.x) {
    case 0: W = Wf1; g = gf1; Nout = 64;  Kin = 70;  KT = 3; NT = 4; dst = dWf1; break;
    case 1: W = Wf2; g = gf2; Nout = 128; Kin = 64;  KT = 2; NT = 8; dst = dWf2; break;
    case 2: W = Wp1; g = gp1; Nout = 64;  Kin = 70;  KT = 3; NT = 4; dst = dWp1; break;
    case 3: W = Wp2; g = gp2; Nout = 128; Kin = 64;  KT = 2; NT = 8; dst = dWp2; break;
    case 4: W = Wkf; g = nullptr; Nout = 128; Kin = 128; KT = 4; NT = 8; dst = dWkf; break;
    case 5: W = Wvf; g = nullptr; Nout = 128; Kin = 128; KT = 4; NT = 8; dst = dWvf; break;
    case 6: W = Wkp; g = nullptr; Nout = 128; Kin = 128; KT = 4; NT = 8; dst = dWkp; break;
    default: W = Wvp; g = nullptr; Nout = 128; Kin = 128; KT = 4; NT = 8; dst = dWvp; break;
  }
  int total = KT * NT * 512;
  float bnscale = rsqrtf(1.f + EPSBN);
  for (int i = threadIdx.x; i < total; i += blockDim.x) {
    int e = i & 15;
    int lane = (i >> 4) & 31;
    int tile = i >> 9;                 // kt*NT + nt
    int nt = tile % NT;
    int kt = tile / NT;
    int n = nt * 16 + (lane & 15);
    int kk = kt * 32 + ((lane >> 4) << 4) + e;
    float val = 0.f;
    if (kk < Kin && n < Nout) {
      val = W[n * Kin + kk];
      if (g) val *= g[n] * bnscale;    // fold BN scale into weights
    }
    dst[i] = (half_t)val;
  }
}

__global__ void colsum_kernel(const float* Wqf, const float* Wqp,
                              float* csf, float* csp) {
  int k = threadIdx.x;                 // 128 threads
  float a = 0.f, b = 0.f;
  for (int c = 0; c < MLP1C; ++c) { a += Wqf[c * MLP1C + k]; b += Wqp[c * MLP1C + k]; }
  csf[k] = a; csp[k] = b;
}

// ------------------------------------------------------------------------ kNN
__global__ void __launch_bounds__(32) knn_kernel(const float* xyz, const int* fps,
                                                 int* knn, float* out_newxyz,
                                                 float* out_fps) {
  __shared__ float dist[NN];
  int bs = blockIdx.x;
  int b = bs / SS, s = bs % SS;
  int lane = threadIdx.x;
  int fi = fps[b * SS + s];
  float q0 = xyz[(b * 3 + 0) * NN + fi];
  float q1 = xyz[(b * 3 + 1) * NN + fi];
  float q2 = xyz[(b * 3 + 2) * NN + fi];
  for (int n = lane; n < NN; n += 32) {
    float d0 = xyz[(b * 3 + 0) * NN + n] - q0;
    float d1 = xyz[(b * 3 + 1) * NN + n] - q1;
    float d2 = xyz[(b * 3 + 2) * NN + n] - q2;
    dist[n] = d0 * d0 + d1 * d1 + d2 * d2;
  }
  __syncthreads();
  for (int r = 0; r < KNS; ++r) {
    float best = 3.4e38f; int bi = NN;
    for (int n = lane; n < NN; n += 32) {
      float v = dist[n];
      if (v < best) { best = v; bi = n; }
    }
#pragma unroll
    for (int m = 16; m >= 1; m >>= 1) {
      float ov = __shfl_xor(best, m, 32);
      int   oi = __shfl_xor(bi, m, 32);
      if (ov < best || (ov == best && oi < bi)) { best = ov; bi = oi; }
    }
    if (lane == 0) knn[bs * KNS + r] = bi;
    dist[bi] = 3.4e38f;                // all lanes agree on bi (same value)
    __syncthreads();
  }
  if (lane < 3) {
    float q = (lane == 0) ? q0 : ((lane == 1) ? q1 : q2);
    out_newxyz[(b * 3 + lane) * SS + s] = q;   // T(new_xyz)
  }
  if (lane == 0) out_fps[b * SS + s] = (float)fi;
}

// -------------------------------------------- fused gather + MLP + max(NS)
__global__ void __launch_bounds__(32) mlp_kernel(
    const float* xyz, const float* points, const float* csa_p,
    const int* fps, const int* knn,
    const half_t* fW1f, const float* bf1, const half_t* fW2f, const float* bf2,
    const half_t* fW1p, const float* bp1, const half_t* fW2p, const float* bp2,
    float* feat_ws, float* pos_ws) {
  __shared__ half_t cat[32 * 96];       // 32 neighbors x (70 -> pad 96) channels
  __shared__ half_t a2[32 * 64];        // layer-1 output
  int bs = blockIdx.x;
  int b = bs / SS, s = bs % SS;
  int lane = threadIdx.x;
  int fi = fps[b * SS + s];
  float q0 = xyz[(b * 3 + 0) * NN + fi];
  float q1 = xyz[(b * 3 + 1) * NN + fi];
  float q2 = xyz[(b * 3 + 2) * NN + fi];
  int nidx = knn[(size_t)bs * KNS + lane];
  float g0 = xyz[(b * 3 + 0) * NN + nidx];
  float g1 = xyz[(b * 3 + 1) * NN + nidx];
  float g2 = xyz[(b * 3 + 2) * NN + nidx];
  int col = lane & 15;
  int hi = (lane >> 4) << 3;

  for (int br = 0; br < 2; ++br) {
    const float* src = br ? csa_p : points;
    const half_t* W1 = br ? fW1p : fW1f;
    const half_t* W2 = br ? fW2p : fW2f;
    const float* B1 = br ? bp1 : bf1;
    const float* B2 = br ? bp2 : bf2;
    float* outw = br ? pos_ws : feat_ws;

    half_t* row = &cat[lane * 96];      // lane owns neighbor `lane`
#pragma unroll 4
    for (int d = 0; d < DD; ++d) row[d] = (half_t)src[(size_t)(b * DD + d) * NN + nidx];
    row[64] = (half_t)(g0 - q0); row[65] = (half_t)(g1 - q1); row[66] = (half_t)(g2 - q2);
    row[67] = (half_t)q0; row[68] = (half_t)q1; row[69] = (half_t)q2;
#pragma unroll
    for (int d = 70; d < 96; ++d) row[d] = (half_t)0.f;
    __syncthreads();

    // GEMM1: [32x96] x [96x64] -> bias+relu -> a2 (f16)
    for (int mt = 0; mt < 2; ++mt) {
      v16h afr[3];
#pragma unroll
      for (int kt = 0; kt < 3; ++kt) afr[kt] = load_a_frag(cat, 96, mt * 16, kt);
#pragma unroll
      for (int nt = 0; nt < 4; ++nt) {
        v8f acc = vzero8();
#pragma unroll
        for (int kt = 0; kt < 3; ++kt) acc = wmma16(afr[kt], load_b_frag(W1, 4, kt, nt), acc);
        int n = nt * 16 + col;
        float bias = B1[n];
#pragma unroll
        for (int r = 0; r < 8; ++r) {
          float v = fmaxf(acc[r] + bias, 0.f);
          a2[(mt * 16 + r + hi) * 64 + n] = (half_t)v;
        }
      }
    }
    __syncthreads();

    // GEMM2: [32x64] x [64x128] -> bias+relu -> max over 32 rows
    float cm[8];
#pragma unroll
    for (int nt = 0; nt < 8; ++nt) cm[nt] = 0.f;
    for (int mt = 0; mt < 2; ++mt) {
      v16h afr[2];
#pragma unroll
      for (int kt = 0; kt < 2; ++kt) afr[kt] = load_a_frag(a2, 64, mt * 16, kt);
#pragma unroll
      for (int nt = 0; nt < 8; ++nt) {
        v8f acc = vzero8();
#pragma unroll
        for (int kt = 0; kt < 2; ++kt) acc = wmma16(afr[kt], load_b_frag(W2, 8, kt, nt), acc);
        float bias = B2[nt * 16 + col];
        float m = 0.f;                 // relu output >= 0
#pragma unroll
        for (int r = 0; r < 8; ++r) m = fmaxf(m, fmaxf(acc[r] + bias, 0.f));
        m = fmaxf(m, __shfl_xor(m, 16, 32));   // merge rows r+8 (other lane half)
        cm[nt] = fmaxf(cm[nt], m);
      }
    }
    if (lane < 16) {
#pragma unroll
      for (int nt = 0; nt < 8; ++nt)
        outw[(size_t)(b * SS + s) * MLP1C + nt * 16 + lane] = cm[nt];
    }
    __syncthreads();
  }
}

// -------------------------------------------------------------- SE attention
__global__ void tot_kernel(const float* x, float* tot) {
  int i = blockIdx.x * blockDim.x + threadIdx.x;   // i in [0, B*S)
  const float* row = x + (size_t)i * MLP1C;
  float t = 0.f;
#pragma unroll 8
  for (int c = 0; c < MLP1C; ++c) t += row[c];
  tot[i] = t;
}

__global__ void __launch_bounds__(128) attn_scale_kernel(
    const float* x, const float* tot, const float* W1, const float* g1,
    const float* b1, const float* W2, float* scale_out) {
  __shared__ float att[MLP1C];
  __shared__ float h[32];
  int b = blockIdx.x;
  int c = threadIdx.x;
  float a = 0.f;
  for (int s = 0; s < SS; ++s)
    a += x[(size_t)(b * SS + s) * MLP1C + c] * tot[b * SS + s];
  att[c] = a;
  __syncthreads();
  if (c < 32) {
    float v = 0.f;
    for (int k = 0; k < MLP1C; ++k) v += att[k] * W1[c * MLP1C + k];
    v = v * (g1[c] * rsqrtf(1.f + EPSBN)) + b1[c];
    h[c] = fmaxf(v, 0.f);
  }
  __syncthreads();
  float v = 0.f;
#pragma unroll
  for (int k = 0; k < 32; ++k) v += h[k] * W2[c * 32 + k];
  scale_out[b * MLP1C + c] = 1.f / (1.f + expf(-v));
}

// --------------------------------------------------------------------- CSA
__global__ void __launch_bounds__(32) csa_kernel(
    const float* feat_ws, const float* pos_ws, const float* scale_f,
    const float* scale_p, const half_t* fWkf, const half_t* fWvf,
    const half_t* fWkp, const half_t* fWvp, const float* colsum_f,
    const float* colsum_p, float* out_base) {
  __shared__ half_t Ap[32 * MLP1C];
  __shared__ half_t Af[32 * MLP1C];
  __shared__ float qsum[32];
  int s0 = blockIdx.x * 32;
  int b = blockIdx.y;
  int br = blockIdx.z;                  // 0: csa_f, 1: csa_po
  int lane = threadIdx.x;
  const float* posx  = br ? feat_ws : pos_ws;   // "position" operand
  const float* featx = br ? pos_ws  : feat_ws;  // "feature" operand
  const float* scPos  = br ? scale_f : scale_p;
  const float* scFeat = br ? scale_p : scale_f;
  const half_t* Bk = br ? fWkp : fWkf;
  const half_t* Bv = br ? fWvp : fWvf;
  const float* cs = br ? colsum_p : colsum_f;
  float* outp = out_base + (size_t)br * (BB * MLP1C * SS);

  size_t rowoff = (size_t)(b * SS + s0 + lane) * MLP1C;
  float qs = 0.f;
  for (int c = 0; c < MLP1C; ++c) {
    float pv = posx[rowoff + c] * scPos[b * MLP1C + c];
    float fv = featx[rowoff + c] * scFeat[b * MLP1C + c];
    Ap[lane * MLP1C + c] = (half_t)pv;
    Af[lane * MLP1C + c] = (half_t)fv;
    qs += pv * cs[c];                   // q.sum(-1) == pos . colsum(Wq)
  }
  qsum[lane] = qs;
  __syncthreads();

  int col = lane & 15;
  int hi = (lane >> 4) << 3;
  for (int mt = 0; mt < 2; ++mt) {
    // k = position @ Wk.T
    v16h apf[4];
#pragma unroll
    for (int kt = 0; kt < 4; ++kt) apf[kt] = load_a_frag(Ap, MLP1C, mt * 16, kt);
    v8f e[8];
#pragma unroll
    for (int nt = 0; nt < 8; ++nt) {
      v8f acc = vzero8();
#pragma unroll
      for (int kt = 0; kt < 4; ++kt) acc = wmma16(apf[kt], load_b_frag(Bk, 8, kt, nt), acc);
      e[nt] = acc;
    }
    // energy = qsum(row) * k ; normalize by row sum; softmax over channels
    float qrow[8];
#pragma unroll
    for (int r = 0; r < 8; ++r) qrow[r] = qsum[mt * 16 + r + hi];
#pragma unroll
    for (int nt = 0; nt < 8; ++nt)
#pragma unroll
      for (int r = 0; r < 8; ++r) e[nt][r] *= qrow[r];
#pragma unroll
    for (int r = 0; r < 8; ++r) {
      float t = 0.f;
#pragma unroll
      for (int nt = 0; nt < 8; ++nt) t += e[nt][r];
      t += __shfl_xor(t, 1, 32); t += __shfl_xor(t, 2, 32);
      t += __shfl_xor(t, 4, 32); t += __shfl_xor(t, 8, 32);
      float inv = 1.f / (1e-9f + t);
#pragma unroll
      for (int nt = 0; nt < 8; ++nt) e[nt][r] *= inv;
      // softmax over 128 channels for this row
      float m = -3.4e38f;
#pragma unroll
      for (int nt = 0; nt < 8; ++nt) m = fmaxf(m, e[nt][r]);
      m = fmaxf(m, __shfl_xor(m, 1, 32)); m = fmaxf(m, __shfl_xor(m, 2, 32));
      m = fmaxf(m, __shfl_xor(m, 4, 32)); m = fmaxf(m, __shfl_xor(m, 8, 32));
      float ps = 0.f;
#pragma unroll
      for (int nt = 0; nt < 8; ++nt) { float p = expf(e[nt][r] - m); e[nt][r] = p; ps += p; }
      ps += __shfl_xor(ps, 1, 32); ps += __shfl_xor(ps, 2, 32);
      ps += __shfl_xor(ps, 4, 32); ps += __shfl_xor(ps, 8, 32);
      float pinv = 1.f / ps;
#pragma unroll
      for (int nt = 0; nt < 8; ++nt) e[nt][r] *= pinv;   // e now holds attn
    }
    // v = feature @ Wv.T ; out = attn*v + feature (channel-first scatter)
    v16h aff[4];
#pragma unroll
    for (int kt = 0; kt < 4; ++kt) aff[kt] = load_a_frag(Af, MLP1C, mt * 16, kt);
#pragma unroll
    for (int nt = 0; nt < 8; ++nt) {
      v8f vacc = vzero8();
#pragma unroll
      for (int kt = 0; kt < 4; ++kt) vacc = wmma16(aff[kt], load_b_frag(Bv, 8, kt, nt), vacc);
#pragma unroll
      for (int r = 0; r < 8; ++r) {
        int row = mt * 16 + r + hi;
        int cc = nt * 16 + col;
        float fval = (float)Af[row * MLP1C + cc];
        outp[((size_t)b * MLP1C + cc) * SS + (s0 + row)] = e[nt][r] * vacc[r] + fval;
      }
    }
  }
}

// ------------------------------------------------------------------- launch
extern "C" void kernel_launch(void* const* d_in, const int* in_sizes, int n_in,
                              void* d_out, int out_size, void* d_ws, size_t ws_size,
                              hipStream_t stream) {
  (void)in_sizes; (void)n_in; (void)out_size; (void)ws_size;
  const float* xyz    = (const float*)d_in[0];
  const float* points = (const float*)d_in[1];
  const float* csa_p  = (const float*)d_in[2];
  const int*   fps    = (const int*)d_in[3];
  const float* Wf1 = (const float*)d_in[4];
  const float* gf1 = (const float*)d_in[5];
  const float* bf1 = (const float*)d_in[6];
  const float* Wf2 = (const float*)d_in[7];
  const float* gf2 = (const float*)d_in[8];
  const float* bf2 = (const float*)d_in[9];
  const float* Wp1 = (const float*)d_in[10];
  const float* gp1 = (const float*)d_in[11];
  const float* bp1 = (const float*)d_in[12];
  const float* Wp2 = (const float*)d_in[13];
  const float* gp2 = (const float*)d_in[14];
  const float* bp2 = (const float*)d_in[15];
  const float* Wqf = (const float*)d_in[16];
  const float* Wkf = (const float*)d_in[17];
  const float* Wvf = (const float*)d_in[18];
  const float* Wqp = (const float*)d_in[19];
  const float* Wkp = (const float*)d_in[20];
  const float* Wvp = (const float*)d_in[21];
  const float* Wa1_1 = (const float*)d_in[22];
  const float* ga1   = (const float*)d_in[23];
  const float* ba1   = (const float*)d_in[24];
  const float* Wa1_2 = (const float*)d_in[25];
  const float* Wa2_1 = (const float*)d_in[26];
  const float* ga2   = (const float*)d_in[27];
  const float* ba2   = (const float*)d_in[28];
  const float* Wa2_2 = (const float*)d_in[29];

  float* out = (float*)d_out;
  float* out_newxyz = out;                                   // [16,3,1024]
  float* out_csa    = out + (size_t)BB * 3 * SS;             // csa_f then csa_po
  float* out_fps    = out + (size_t)BB * 3 * SS + 2ull * BB * MLP1C * SS;

  char* ws = (char*)d_ws;
  size_t o = 0;
  auto alloc = [&](size_t bytes) -> char* {
    char* p = ws + o;
    o = (o + bytes + 255) & ~(size_t)255;
    return p;
  };
  int*    knn      = (int*)alloc((size_t)BB * SS * KNS * 4);         // 2 MB
  float*  feat_ws  = (float*)alloc((size_t)BB * SS * MLP1C * 4);     // 8 MB
  float*  pos_ws   = (float*)alloc((size_t)BB * SS * MLP1C * 4);     // 8 MB
  float*  tot_f    = (float*)alloc((size_t)BB * SS * 4);
  float*  tot_p    = (float*)alloc((size_t)BB * SS * 4);
  float*  scale_f  = (float*)alloc((size_t)BB * MLP1C * 4);
  float*  scale_p  = (float*)alloc((size_t)BB * MLP1C * 4);
  float*  colsum_f = (float*)alloc(MLP1C * 4);
  float*  colsum_p = (float*)alloc(MLP1C * 4);
  half_t* dWf1 = (half_t*)alloc(3 * 4 * 512 * 2);
  half_t* dWf2 = (half_t*)alloc(2 * 8 * 512 * 2);
  half_t* dWp1 = (half_t*)alloc(3 * 4 * 512 * 2);
  half_t* dWp2 = (half_t*)alloc(2 * 8 * 512 * 2);
  half_t* dWkf = (half_t*)alloc(4 * 8 * 512 * 2);
  half_t* dWvf = (half_t*)alloc(4 * 8 * 512 * 2);
  half_t* dWkp = (half_t*)alloc(4 * 8 * 512 * 2);
  half_t* dWvp = (half_t*)alloc(4 * 8 * 512 * 2);

  prep_frags_kernel<<<8, 256, 0, stream>>>(Wf1, gf1, Wf2, gf2, Wp1, gp1, Wp2, gp2,
                                           Wkf, Wvf, Wkp, Wvp,
                                           dWf1, dWf2, dWp1, dWp2,
                                           dWkf, dWvf, dWkp, dWvp);
  colsum_kernel<<<1, 128, 0, stream>>>(Wqf, Wqp, colsum_f, colsum_p);

  knn_kernel<<<BB * SS, 32, 0, stream>>>(xyz, fps, knn, out_newxyz, out_fps);

  mlp_kernel<<<BB * SS, 32, 0, stream>>>(xyz, points, csa_p, fps, knn,
                                         dWf1, bf1, dWf2, bf2,
                                         dWp1, bp1, dWp2, bp2,
                                         feat_ws, pos_ws);

  tot_kernel<<<(BB * SS) / 256, 256, 0, stream>>>(feat_ws, tot_f);
  tot_kernel<<<(BB * SS) / 256, 256, 0, stream>>>(pos_ws, tot_p);
  // feat uses Wa2_*, pos uses Wa1_* (per reference)
  attn_scale_kernel<<<BB, 128, 0, stream>>>(feat_ws, tot_f, Wa2_1, ga2, ba2, Wa2_2, scale_f);
  attn_scale_kernel<<<BB, 128, 0, stream>>>(pos_ws, tot_p, Wa1_1, ga1, ba1, Wa1_2, scale_p);

  csa_kernel<<<dim3(SS / 32, BB, 2), 32, 0, stream>>>(
      feat_ws, pos_ws, scale_f, scale_p, dWkf, dWvf, dWkp, dWvp,
      colsum_f, colsum_p, out_csa);
}
